// WaveNOARZ_64441689309841
// MI455X (gfx1250) — matrix operand order repaired and checked by reading, not compiled, not run on hardware
//
#include <hip/hip_runtime.h>

#define HIDD 128
#define NHEAD 8
#define DHH 16
#define KPP 64
#define NBB 8
#define NTT 50
#define NXX 256
#define FFD 512

typedef __attribute__((ext_vector_type(16))) _Float16 v16h;
typedef __attribute__((ext_vector_type(8)))  _Float16 v8h;
typedef __attribute__((ext_vector_type(4)))  _Float16 v4h;
typedef __attribute__((ext_vector_type(8)))  float    v8f;

#define LOG2E 1.44269504088896f

__device__ __forceinline__ v8f wmma16(v16h a, v16h b, v8f c) {
  return __builtin_amdgcn_wmma_f32_16x16x32_f16(false, a, false, b, (short)0, c, false, false);
}
// branch-free fast transcendentals on raw HW ops (v_exp_f32 / v_rcp_f32)
__device__ __forceinline__ float fexp(float x) {      // e^x
  return __builtin_amdgcn_exp2f(x * LOG2E);
}
__device__ __forceinline__ float frcp(float x) { return __builtin_amdgcn_rcpf(x); }
__device__ __forceinline__ float sigmoid_f(float z) { // 1/(1+e^-z), saturates cleanly
  return frcp(1.f + __builtin_amdgcn_exp2f(-z * LOG2E));
}
// tanh-approx GELU == x * sigmoid(1.5957692*(x + 0.044715 x^3))
__device__ __forceinline__ float gelu_f(float x) {
  return x * sigmoid_f(1.595769122f * (x + 0.044715f * x * x * x));
}
// A-fragment (16x32 f16): per lane two contiguous 8-half runs at K = 8*hi and 16+8*hi.
__device__ __forceinline__ v16h loadA(const _Float16* p, int hi) {
  const v8h a0 = *(const v8h*)(p + 8 * hi);
  const v8h a1 = *(const v8h*)(p + 16 + 8 * hi);
  v16h r;
#pragma unroll
  for (int i = 0; i < 8; ++i) { r[i] = a0[i]; r[8 + i] = a1[i]; }
  return r;
}

// ---------------------------------------------------------------------------
// GEMM: C[M,N] = act(A16[M,Kd] @ W16[Kd,N] + bias[N]), optional f16 shadow C16.
// M%(16*MT)==0, N%128==0, Kd%32==0. Wave: 16*MT rows x 16 cols; one B load
// feeds MT WMMAs.
// ---------------------------------------------------------------------------
template <int ACT, int MT, typename TC>
__global__ void gemm_kernel(const _Float16* __restrict__ A, const _Float16* __restrict__ W,
                            const float* __restrict__ bias, TC* __restrict__ C,
                            _Float16* __restrict__ C16, int M, int N, int Kd) {
  const int lane = threadIdx.x & 31, wv = threadIdx.x >> 5;
  const int hi = lane >> 4, lm = lane & 15;
  const int m0 = blockIdx.x * (16 * MT);
  const int n0 = blockIdx.y * 128 + wv * 16;
  const _Float16* Ar[MT];
#pragma unroll
  for (int t = 0; t < MT; ++t) Ar[t] = A + (long)(m0 + 16 * t + lm) * Kd;
  v8f acc[MT];
#pragma unroll
  for (int t = 0; t < MT; ++t) acc[t] = (v8f){};
  for (int k0 = 0; k0 < Kd; k0 += 32) {
    v16h b;
#pragma unroll
    for (int i = 0; i < 8; ++i) { b[i] = W[(long)(k0 + 8 * hi + i) * N + n0 + lm]; b[8 + i] = W[(long)(k0 + 16 + 8 * hi + i) * N + n0 + lm]; }
    if (k0 + 32 < Kd) __builtin_prefetch(W + (long)(k0 + 32 + lane) * N + n0, 0, 1);
#pragma unroll
    for (int t = 0; t < MT; ++t) {
      v16h a = loadA(Ar[t] + k0, hi);
      acc[t] = wmma16(a, b, acc[t]);
    }
  }
  const float bv = bias[n0 + lm];
#pragma unroll
  for (int t = 0; t < MT; ++t) {
    TC* Cc = C + (long)(m0 + 16 * t) * N + n0 + lm;
#pragma unroll
    for (int r = 0; r < 8; ++r) {
      float v = acc[t][r] + bv;
      if (ACT == 1) v = gelu_f(v);
      if (ACT == 2) v = fmaxf(v, 0.f);
      Cc[(long)(r + 8 * hi) * N] = (TC)v;
      if (C16) C16[(long)(m0 + 16 * t + r + 8 * hi) * N + n0 + lm] = (_Float16)v;
    }
  }
}

// ---------------------------------------------------------------------------
// Fused FFN: Y16 = gelu(X16@W1+b1)@W2 + b2 ; 128->512->128, 16-row tile.
// A-fragments hoisted; W2 prefetched behind phase-1 WMMAs.
// ---------------------------------------------------------------------------
__global__ void ffn_kernel(const _Float16* __restrict__ X, const _Float16* __restrict__ W1,
                           const float* __restrict__ b1, const _Float16* __restrict__ W2,
                           const float* __restrict__ b2, _Float16* __restrict__ Y, int M) {
  __shared__ _Float16 hid[16 * FFD];  // 16KB
  const int lane = threadIdx.x & 31, wv = threadIdx.x >> 5;
  const int hi = lane >> 4, lm = lane & 15;
  const int m0 = blockIdx.x * 16;
  const _Float16* Xrow = X + (long)(m0 + lm) * HIDD;
  v16h af[4];
#pragma unroll
  for (int j = 0; j < 4; ++j) af[j] = loadA(Xrow + 32 * j, hi);
#pragma unroll
  for (int nt = 0; nt < 4; ++nt) {
    const int n0 = wv * 64 + nt * 16;
    v8f acc = {};
#pragma unroll
    for (int j = 0; j < 4; ++j) {
      v16h b;
#pragma unroll
      for (int i = 0; i < 8; ++i) { b[i] = W1[(long)(32 * j + 8 * hi + i) * FFD + n0 + lm]; b[8 + i] = W1[(long)(32 * j + 16 + 8 * hi + i) * FFD + n0 + lm]; }
      acc = wmma16(af[j], b, acc);
    }
    __builtin_prefetch(W2 + (long)(nt * 128 + lane) * HIDD + wv * 16, 0, 1);
    const float bv = b1[n0 + lm];
#pragma unroll
    for (int r = 0; r < 8; ++r)
      hid[(r + 8 * hi) * FFD + n0 + lm] = (_Float16)gelu_f(acc[r] + bv);
  }
  __syncthreads();
  const int n0 = wv * 16;
  v8f acc = {};
#pragma unroll
  for (int k0 = 0; k0 < FFD; k0 += 32) {
    v16h a = loadA(&hid[lm * FFD + k0], hi);
    v16h b;
#pragma unroll
    for (int i = 0; i < 8; ++i) { b[i] = W2[(long)(k0 + 8 * hi + i) * HIDD + n0 + lm]; b[8 + i] = W2[(long)(k0 + 16 + 8 * hi + i) * HIDD + n0 + lm]; }
    acc = wmma16(a, b, acc);
  }
  const float bv = b2[n0 + lm];
#pragma unroll
  for (int r = 0; r < 8; ++r)
    Y[(long)(m0 + r + 8 * hi) * HIDD + n0 + lm] = (_Float16)(acc[r] + bv);
}

// ---------------------------------------------------------------------------
// Residual + LayerNorm over H=128. X is f16, R (f32) may be null,
// Y32 (f32 residual stream) may be null, Y16 (f16 shadow) always written.
// ---------------------------------------------------------------------------
__global__ void resln_kernel(const _Float16* __restrict__ X, const float* __restrict__ R,
                             const float* __restrict__ g, const float* __restrict__ b,
                             float* __restrict__ Y32, _Float16* __restrict__ Y16, int M) {
  const int lane = threadIdx.x & 31, wv = threadIdx.x >> 5;
  const long row = (long)blockIdx.x * 8 + wv;
  if (row >= M) return;
  v4h xv = *(const v4h*)(X + row * HIDD + lane * 4);
  float v[4];
#pragma unroll
  for (int i = 0; i < 4; ++i) v[i] = (float)xv[i];
  if (R) {
    float4 rv = *(const float4*)(R + row * HIDD + lane * 4);
    v[0] += rv.x; v[1] += rv.y; v[2] += rv.z; v[3] += rv.w;
  }
  float s = v[0] + v[1] + v[2] + v[3];
#pragma unroll
  for (int off = 16; off; off >>= 1) s += __shfl_xor(s, off);
  const float mean = s * (1.f / 128.f);
  float var = 0.f;
#pragma unroll
  for (int i = 0; i < 4; ++i) { float d = v[i] - mean; var += d * d; }
#pragma unroll
  for (int off = 16; off; off >>= 1) var += __shfl_xor(var, off);
  const float inv = rsqrtf(var * (1.f / 128.f) + 1e-5f);
  const int c = lane * 4;
  float4 gv = *(const float4*)(g + c);
  float4 bv = *(const float4*)(b + c);
  float o[4];
  o[0] = (v[0] - mean) * inv * gv.x + bv.x;
  o[1] = (v[1] - mean) * inv * gv.y + bv.y;
  o[2] = (v[2] - mean) * inv * gv.z + bv.z;
  o[3] = (v[3] - mean) * inv * gv.w + bv.w;
  if (Y32) {
    float4 ov = {o[0], o[1], o[2], o[3]};
    *(float4*)(Y32 + row * HIDD + c) = ov;
  }
  v4h oh;
#pragma unroll
  for (int i = 0; i < 4; ++i) oh[i] = (_Float16)o[i];
  *(v4h*)(Y16 + row * HIDD + c) = oh;
}

// ---------------------------------------------------------------------------
// Attention (f16): O = softmax(QK^T/4 + bias + mask) @ V ; Lk=64, dh=16,
// 8 heads, wave = head; S^T = K@Q^T so k-softmax is in-lane + xor-16 shuffle.
// ---------------------------------------------------------------------------
__global__ void attn_kernel(const _Float16* __restrict__ Qp, const _Float16* __restrict__ Kp,
                            const _Float16* __restrict__ Vp, const float* __restrict__ bias,
                            const float* __restrict__ pmask, int pm_div,
                            _Float16* __restrict__ O, int Nb, int Lq) {
  __shared__ _Float16 pl[NHEAD * 16 * KPP];  // per head, q-major [q][kp], 16KB
  const int lane = threadIdx.x & 31, head = threadIdx.x >> 5;
  const int hi = lane >> 4, lm = lane & 15;
  const int lqt = Lq >> 4;
  const int n = blockIdx.x / lqt;
  const int q0 = (blockIdx.x % lqt) * 16;
  const _Float16* Qb = Qp + (long)n * Lq * HIDD + head * DHH;
  const _Float16* Kb = Kp + (long)n * KPP * HIDD + head * DHH;
  const _Float16* Vb = Vp + (long)n * KPP * HIDD + head * DHH;

  v16h bq;  // Q^T B-fragment (col = q = lane%16, elems 0..7 = dh 8*hi+i; halves 8..15 zero pad)
#pragma unroll
  for (int h = 0; h < 16; ++h) {
    bq[h] = (h < 8) ? Qb[(long)(q0 + lm) * HIDD + 8 * hi + h] : (_Float16)0.f;
  }
  v8f s[4];
#pragma unroll
  for (int c = 0; c < 4; ++c) {
    v16h a = {};  // halves 8..15 stay zero (dh pad)
    v8h k8 = *(const v8h*)(Kb + (16 * c + lm) * HIDD + 8 * hi);
#pragma unroll
    for (int i = 0; i < 8; ++i) a[i] = k8[i];
    v8f z = {};
    s[c] = wmma16(a, bq, z);
  }
  const float* prow = pmask ? pmask + (long)(n / pm_div) * KPP : nullptr;
  const float* brow = bias ? bias + ((long)n * Lq + q0 + lm) * KPP : nullptr;
  float mx = -1e30f;
#pragma unroll
  for (int c = 0; c < 4; ++c)
#pragma unroll
    for (int r = 0; r < 8; ++r) {
      int kp = 16 * c + r + 8 * hi;
      float v = s[c][r] * 0.25f;
      if (brow) v += brow[kp];
      if (prow && prow[kp] < 0.5f) v = -1e9f;
      s[c][r] = v;
      mx = fmaxf(mx, v);
    }
  mx = fmaxf(mx, __shfl_xor(mx, 16));
  float sum = 0.f;
#pragma unroll
  for (int c = 0; c < 4; ++c)
#pragma unroll
    for (int r = 0; r < 8; ++r) {
      float e = __builtin_amdgcn_exp2f((s[c][r] - mx) * LOG2E);  // arg <= 0: raw exp safe
      s[c][r] = e;
      sum += e;
    }
  sum += __shfl_xor(sum, 16);
  const float rinv = frcp(sum);
  _Float16* plh = pl + head * (16 * KPP);
#pragma unroll
  for (int c = 0; c < 4; ++c) {
    v8h pk;
#pragma unroll
    for (int r = 0; r < 8; ++r) pk[r] = (_Float16)(s[c][r] * rinv);
    *(v8h*)&plh[lm * KPP + 16 * c + 8 * hi] = pk;
  }
  __syncthreads();
  v8f acc = {};
#pragma unroll
  for (int ac = 0; ac < 2; ++ac) {
    v16h a = loadA(&plh[lm * KPP + 32 * ac], hi);
    v16h b;
#pragma unroll
    for (int i = 0; i < 8; ++i) { b[i] = Vb[(long)(32 * ac + 8 * hi + i) * HIDD + lm]; b[8 + i] = Vb[(long)(32 * ac + 16 + 8 * hi + i) * HIDD + lm]; }
    acc = wmma16(a, b, acc);
  }
#pragma unroll
  for (int r = 0; r < 8; ++r)
    O[((long)n * Lq + q0 + r + 8 * hi) * HIDD + head * DHH + lm] = (_Float16)acc[r];
}

// ---------------------------------------------------------------------------
// Elementwise / feature kernels
// ---------------------------------------------------------------------------
__device__ __forceinline__ void write_fourier8h(float v, _Float16* o) {
  o[0] = (_Float16)v;
#pragma unroll
  for (int j = 0; j < 8; ++j) {
    float a = v * (3.14159265358979f * (float)(1 << j));
    o[1 + j] = (_Float16)__sinf(a);
    o[9 + j] = (_Float16)__cosf(a);
  }
}

__global__ void k_w2h(const float* __restrict__ src, _Float16* __restrict__ dst,
                      int Kd, int Kdp, int N) {
  int i = blockIdx.x * blockDim.x + threadIdx.x;
  if (i >= Kdp * N) return;
  int r = i / N;
  dst[i] = (r < Kd) ? (_Float16)src[i] : (_Float16)0.f;
}

__global__ void k_feats(const float* xs, const float* ks, const float* ksv, _Float16* f) {
  int i = blockIdx.x * blockDim.x + threadIdx.x;
  if (i >= NBB * KPP) return;
  int b = i / KPP, k = i % KPP;
  float xl = xs[b * (KPP + 1) + k], xr = xs[b * (KPP + 1) + k + 1];
  float kv = ks[i], vv = ksv[i];
  _Float16* o = f + i * 64;
  write_fourier8h(xl, o);
  write_fourier8h(xr, o + 17);
  o[34] = (_Float16)kv; o[35] = (_Float16)vv; o[36] = (_Float16)kv;
  o[37] = (_Float16)(vv + kv); o[38] = (_Float16)(vv - kv); o[39] = (_Float16)vv;
#pragma unroll
  for (int j = 40; j < 64; ++j) o[j] = (_Float16)0.f;
}

__global__ void k_tf(const float* tcg, _Float16* tf) {
  int i = blockIdx.x * blockDim.x + threadIdx.x;
  if (i >= NBB * NTT) return;
  _Float16* o = tf + i * 32;
  write_fourier8h(tcg[(long)i * NXX], o);
#pragma unroll
  for (int j = 17; j < 32; ++j) o[j] = (_Float16)0.f;
}

__global__ void k_qf(const float* tcg, const float* xcg, _Float16* qf) {
  int i = blockIdx.x * blockDim.x + threadIdx.x;
  if (i >= NBB * NTT * NXX) return;
  _Float16* o = qf + (long)i * 64;
  write_fourier8h(tcg[i], o);
  write_fourier8h(xcg[i], o + 17);
#pragma unroll
  for (int j = 34; j < 64; ++j) o[j] = (_Float16)0.f;
}

// seg_t = (seg*pm)*(1+gam) + bet  -> f32 residual + f16 shadow
__global__ void k_film(const float* seg, const float* gb, const float* pm,
                       float* kv, _Float16* kv16) {
  long i = (long)blockIdx.x * blockDim.x + threadIdx.x;
  if (i >= (long)NBB * NTT * KPP * HIDD) return;
  int h = i & (HIDD - 1);
  int k = (i >> 7) & (KPP - 1);
  int nt = (int)(i >> 13);
  int b = nt / NTT;
  float sv = seg[((long)b * KPP + k) * HIDD + h] * pm[b * KPP + k];
  float g = gb[(long)nt * 256 + h];
  float be = gb[(long)nt * 256 + 128 + h];
  float v = sv * (1.f + g) + be;
  kv[i] = v;
  kv16[i] = (_Float16)v;
}

__global__ void k_maskmul(float* kv, _Float16* kv16, const float* pm) {
  long i = (long)blockIdx.x * blockDim.x + threadIdx.x;
  if (i >= (long)NBB * NTT * KPP * HIDD) return;
  int k = (i >> 7) & (KPP - 1);
  int b = (int)(i >> 13) / NTT;
  float v = kv[i] * pm[b * KPP + k];
  kv[i] = v;
  kv16[i] = (_Float16)v;
}

__global__ void k_bias(const float* xs, const float* ks, const float* ksv, const float* pm,
                       const float* tcg, const float* xcg, const float* bs, const float* ds,
                       float* cb) {
  long i = (long)blockIdx.x * blockDim.x + threadIdx.x;
  if (i >= (long)NBB * NTT * NXX * KPP) return;
  int k = i & 63;
  int x = (i >> 6) & 255;
  int t = (int)(i >> 14) % NTT;
  int b = (int)(i / ((long)64 * 256 * NTT));
  long row = ((long)b * NTT + t) * NXX + x;
  float tcv = tcg[row], xgv = xcg[row];
  float kk = ks[b * KPP + k], vv = ksv[b * KPP + k];
  float xc = 0.5f * (xs[b * (KPP + 1) + k] + xs[b * (KPP + 1) + k + 1]);
  float d1 = fabsf(xgv - xc - (vv - kk) * tcv);
  float d2 = fabsf(xgv - xc - vv * tcv);
  float damp = sigmoid_f(ds[0] * (1.f - tcv));
  cb[i] = -bs[0] * fminf(d1, d2) * damp * pm[b * KPP + k];
}

// head.l2 (128 -> 2) + transposed store into d_out (B,2,NT,NX)
__global__ void k_head2(const _Float16* __restrict__ X, const float* __restrict__ w,
                        const float* __restrict__ b2, float* __restrict__ dout) {
  const int lane = threadIdx.x & 31, wv = threadIdx.x >> 5;
  const long row = (long)blockIdx.x * 8 + wv;
  if (row >= (long)NBB * NTT * NXX) return;
  v4h xv = *(const v4h*)(X + row * HIDD + lane * 4);
  const int c = lane * 4;
  float x0 = (float)xv[0], x1 = (float)xv[1], x2 = (float)xv[2], x3 = (float)xv[3];
  float s0 = x0 * w[c * 2] + x1 * w[c * 2 + 2] + x2 * w[c * 2 + 4] + x3 * w[c * 2 + 6];
  float s1 = x0 * w[c * 2 + 1] + x1 * w[c * 2 + 3] + x2 * w[c * 2 + 5] + x3 * w[c * 2 + 7];
#pragma unroll
  for (int off = 16; off; off >>= 1) { s0 += __shfl_xor(s0, off); s1 += __shfl_xor(s1, off); }
  if (lane == 0) {
    int b = (int)(row / (NTT * NXX));
    int r = (int)(row % (NTT * NXX));
    dout[(long)(b * 2 + 0) * (NTT * NXX) + r] = s0 + b2[0];
    dout[(long)(b * 2 + 1) * (NTT * NXX) + r] = s1 + b2[1];
  }
}

// ---------------------------------------------------------------------------
// Host-side orchestration
// ---------------------------------------------------------------------------
struct AttnPH {
  const _Float16 *wq, *wk, *wv, *wo, *f1, *f2;
  const float *wqb, *wkb, *wvb, *wob, *f1b, *f2b, *ln1g, *ln1b, *ln2g, *ln2b;
};

static const _Float16* cvtW(hipStream_t st, const float* w, int Kd, int Kdp, int N,
                            _Float16*& pool) {
  _Float16* dst = pool;
  pool += (size_t)Kdp * N;
  int n = Kdp * N;
  k_w2h<<<(n + 255) / 256, 256, 0, st>>>(w, dst, Kd, Kdp, N);
  return dst;
}

static AttnPH layerH(hipStream_t st, const float* const* P, int s, _Float16*& pool) {
  AttnPH a;
  a.f1b = P[s + 0];  a.f1 = cvtW(st, P[s + 1], 128, 128, 512, pool);
  a.f2b = P[s + 2];  a.f2 = cvtW(st, P[s + 3], 512, 512, 128, pool);
  a.ln1b = P[s + 4]; a.ln1g = P[s + 5]; a.ln2b = P[s + 6]; a.ln2g = P[s + 7];
  a.wkb = P[s + 8];  a.wk = cvtW(st, P[s + 9], 128, 128, 128, pool);
  a.wob = P[s + 10]; a.wo = cvtW(st, P[s + 11], 128, 128, 128, pool);
  a.wqb = P[s + 12]; a.wq = cvtW(st, P[s + 13], 128, 128, 128, pool);
  a.wvb = P[s + 14]; a.wv = cvtW(st, P[s + 15], 128, 128, 128, pool);
  return a;
}

template <int ACT, int MT, typename TC>
static inline void gemmT(hipStream_t st, const _Float16* A, const _Float16* W,
                         const float* bias, TC* C, _Float16* C16, int M, int N, int Kd) {
  dim3 g(M / (16 * MT), N / 128);
  gemm_kernel<ACT, MT, TC><<<g, 256, 0, st>>>(A, W, bias, C, C16, M, N, Kd);
}

// One transformer block. Xres/X16: residual stream + f16 shadow (updated).
// KV16in: f16 source of K/V projections (== X16 for encoder blocks).
static void run_block(hipStream_t st, float* Xres, _Float16* X16, int Mq,
                      const _Float16* KV16in, int Mkv, int Nb, int Lq,
                      const AttnPH& p, const float* bias, const float* pmask, int pmdiv,
                      _Float16* Qh, _Float16* Kh, _Float16* Vh, _Float16* Ah, _Float16* Th) {
  gemmT<0, 4, _Float16>(st, X16, p.wq, p.wqb, Qh, nullptr, Mq, HIDD, HIDD);
  gemmT<0, 4, _Float16>(st, KV16in, p.wk, p.wkb, Kh, nullptr, Mkv, HIDD, HIDD);
  gemmT<0, 4, _Float16>(st, KV16in, p.wv, p.wvb, Vh, nullptr, Mkv, HIDD, HIDD);
  attn_kernel<<<Nb * (Lq / 16), 256, 0, st>>>(Qh, Kh, Vh, bias, pmask, pmdiv, Ah, Nb, Lq);
  gemmT<0, 4, _Float16>(st, Ah, p.wo, p.wob, Th, nullptr, Mq, HIDD, HIDD);
  resln_kernel<<<(Mq + 7) / 8, 256, 0, st>>>(Th, Xres, p.ln1g, p.ln1b, Xres, X16, Mq);
  ffn_kernel<<<Mq / 16, 256, 0, st>>>(X16, p.f1, p.f1b, p.f2, p.f2b, Th, Mq);
  resln_kernel<<<(Mq + 7) / 8, 256, 0, st>>>(Th, Xres, p.ln2g, p.ln2b, Xres, X16, Mq);
}

extern "C" void kernel_launch(void* const* d_in, const int* in_sizes, int n_in,
                              void* d_out, int out_size, void* d_ws, size_t ws_size,
                              hipStream_t stream) {
  (void)in_sizes; (void)n_in; (void)out_size; (void)ws_size;
  const float* xs  = (const float*)d_in[0];
  const float* ks  = (const float*)d_in[1];
  const float* ksv = (const float*)d_in[2];
  const float* pm  = (const float*)d_in[3];
  const float* tcg = (const float*)d_in[4];
  const float* xcg = (const float*)d_in[5];
  const float* P[102];
  for (int i = 0; i < 102; ++i) P[i] = (const float*)d_in[6 + i];
  // param leaves (JAX pytree / sorted-key order):
  // 0 bias_scale | 1..16 cross_attn0 | 17..32 cross_attn1 | 33..48 cross_seg0
  // 49 damping   | 50..53 head | 54..59 query | 60..65 seg | 66..81 self0
  // 82..97 self1 | 98..101 tc

  // ---- workspace carve ----
  const size_t BIG = 13107200;  // 102400*128
  float* f32p = (float*)d_ws;
  float* QRES = f32p;                  // query residual stream (f32)
  float* KVr  = QRES + BIG;            // cross kv residual stream (25600*128)
  float* SEG  = KVr + 3276800;         // 512*128
  float* GB   = SEG + 65536;           // 400*256
  _Float16* h16 = (_Float16*)(GB + 102400);
  _Float16* Q16  = h16;                // f16 shadow of QRES
  _Float16* QH   = Q16 + BIG;          // Q projection
  _Float16* AH   = QH + BIG;           // attention out
  _Float16* TXh  = AH + BIG;           // wo-out / ffn-out scratch
  _Float16* Mh   = TXh + BIG;          // query-MLP / head scratch
  _Float16* KV16 = Mh + BIG;           // f16 shadow of KVr
  _Float16* KH   = KV16 + 3276800;     // K projection
  _Float16* VH   = KH + 3276800;       // V projection
  _Float16* QF16 = VH + 3276800;       // 102400*64
  _Float16* FE16 = QF16 + 6553600;     // 512*64
  _Float16* SEG16= FE16 + 32768;       // 512*128
  _Float16* T0h  = SEG16 + 65536;      // 512*128
  _Float16* TF16 = T0h + 65536;        // 400*32
  _Float16* TH16 = TF16 + 12800;       // 400*128
  _Float16* wpool= TH16 + 51200;       // f16 weight pool (~1.1M halves)

  // ---- weight conversion (f16, K-padded) ----
  const AttnPH ca0 = layerH(stream, P, 1, wpool);
  const AttnPH ca1 = layerH(stream, P, 17, wpool);
  const AttnPH cs0 = layerH(stream, P, 33, wpool);
  const AttnPH s0  = layerH(stream, P, 66, wpool);
  const AttnPH s1  = layerH(stream, P, 82, wpool);
  const _Float16* Wsl1 = cvtW(stream, P[61], 40, 64, 128, wpool);
  const _Float16* Wsl2 = cvtW(stream, P[63], 128, 128, 128, wpool);
  const _Float16* Wtl1 = cvtW(stream, P[99], 17, 32, 128, wpool);
  const _Float16* Wtl2 = cvtW(stream, P[101], 128, 128, 256, wpool);
  const _Float16* Wql1 = cvtW(stream, P[55], 34, 64, 128, wpool);
  const _Float16* Wql2 = cvtW(stream, P[57], 128, 128, 128, wpool);
  const _Float16* Whl1 = cvtW(stream, P[51], 128, 128, 128, wpool);

  // ---- segment encoder (M = 512) ----
  k_feats<<<2, 256, 0, stream>>>(xs, ks, ksv, FE16);
  gemmT<1, 4, _Float16>(stream, FE16, Wsl1, P[60], T0h, nullptr, 512, HIDD, 64);
  resln_kernel<<<64, 256, 0, stream>>>(T0h, nullptr, P[65], P[64], nullptr, T0h, 512);
  gemmT<0, 4, float>(stream, T0h, Wsl2, P[62], SEG, SEG16, 512, HIDD, HIDD);
  run_block(stream, SEG, SEG16, 512, SEG16, 512, NBB, KPP, s0, nullptr, pm, 1,
            QH, KH, VH, AH, TXh);
  run_block(stream, SEG, SEG16, 512, SEG16, 512, NBB, KPP, s1, nullptr, pm, 1,
            QH, KH, VH, AH, TXh);

  // ---- time conditioning + FiLM -> KVr/KV16 (M = 25600) ----
  k_tf<<<2, 256, 0, stream>>>(tcg, TF16);
  gemmT<1, 1, _Float16>(stream, TF16, Wtl1, P[98], TH16, nullptr, 400, HIDD, 32);
  gemmT<0, 1, float>(stream, TH16, Wtl2, P[100], GB, nullptr, 400, 256, HIDD);
  k_film<<<(3276800 + 255) / 256, 256, 0, stream>>>(SEG, GB, pm, KVr, KV16);
  run_block(stream, KVr, KV16, 25600, KV16, 25600, 400, KPP, cs0, nullptr, pm, NTT,
            QH, KH, VH, AH, TXh);
  k_maskmul<<<(3276800 + 255) / 256, 256, 0, stream>>>(KVr, KV16, pm);

  // ---- query MLP (M = 102400) ----
  k_qf<<<(102400 + 255) / 256, 256, 0, stream>>>(tcg, xcg, QF16);
  gemmT<1, 4, _Float16>(stream, QF16, Wql1, P[54], Mh, nullptr, 102400, HIDD, 64);
  resln_kernel<<<12800, 256, 0, stream>>>(Mh, nullptr, P[59], P[58], nullptr, Mh, 102400);
  gemmT<0, 4, float>(stream, Mh, Wql2, P[56], QRES, Q16, 102400, HIDD, HIDD);

  // ---- char_bias straight into d_out (output #2, reused as attn bias) ----
  float* CB = (float*)d_out + (size_t)NBB * 2 * NTT * NXX;  // +204800
  k_bias<<<(6553600 + 255) / 256, 256, 0, stream>>>(xs, ks, ksv, pm, tcg, xcg, P[0], P[49], CB);

  // ---- cross-attention blocks (Mq = 102400, Lq = 256) ----
  run_block(stream, QRES, Q16, 102400, KV16, 25600, 400, NXX, ca0, CB, pm, NTT,
            QH, KH, VH, AH, TXh);
  run_block(stream, QRES, Q16, 102400, KV16, 25600, 400, NXX, ca1, CB, pm, NTT,
            QH, KH, VH, AH, TXh);

  // ---- head ----
  gemmT<2, 4, _Float16>(stream, Q16, Whl1, P[50], Mh, nullptr, 102400, HIDD, HIDD);
  k_head2<<<12800, 256, 0, stream>>>(Mh, P[53], P[52], (float*)d_out);
}
